// MoELayer_28484223107421
// MI455X (gfx1250) — matrix operand order repaired and checked by reading, not compiled, not run on hardware
//
#include <hip/hip_runtime.h>
#include <hip/hip_bf16.h>
#include <math.h>

// ---------------- problem constants ----------------
#define N_TOK 8192
#define DDIM  1024
#define FDIM  4096
#define NEXP  8
#define TM    64             // tokens per FFN block
#define MT    4              // 16-row M-subtiles per block (TM/16)
#define FB    256            // F-chunk width processed per outer iteration
#define XPITCH (DDIM + 8)    // LDS pitch in ushorts (bank-conflict padding)
#define HPITCH (FB + 8)

typedef __attribute__((ext_vector_type(16))) __bf16       v16bf;
typedef __attribute__((ext_vector_type(8)))  float        v8f;
typedef __attribute__((ext_vector_type(8)))  unsigned int v8u;

// ---------------- helpers ----------------
__device__ __forceinline__ unsigned short f2bf(float f) {
  unsigned int u = __float_as_uint(f);
  u = u + 0x7FFFu + ((u >> 16) & 1u);      // round-to-nearest-even
  return (unsigned short)(u >> 16);
}

__device__ __forceinline__ v16bf make_frag(uint4 a, uint4 b) {
  v8u u;
  u[0] = a.x; u[1] = a.y; u[2] = a.z; u[3] = a.w;
  u[4] = b.x; u[5] = b.y; u[6] = b.z; u[7] = b.w;
  return __builtin_bit_cast(v16bf, u);
}

// ---------------- kernel 0: zero routing counters ----------------
__global__ void init_kernel(int* counts) {
  if (threadIdx.x < NEXP) counts[threadIdx.x] = 0;
}

// ---------------- kernel 1: f32 -> bf16 conversion (vectorized x4) ----------------
__global__ void cvt_bf16_kernel(const float* __restrict__ src,
                                unsigned short* __restrict__ dst, int n4) {
  int i = blockIdx.x * blockDim.x + threadIdx.x;
  if (i >= n4) return;
  float4 v = ((const float4*)src)[i];
  unsigned int lo = (unsigned int)f2bf(v.x) | ((unsigned int)f2bf(v.y) << 16);
  unsigned int hi = (unsigned int)f2bf(v.z) | ((unsigned int)f2bf(v.w) << 16);
  ((uint2*)dst)[i] = make_uint2(lo, hi);
}

// ---------------- kernel 2: router (1 wave32 per token) ----------------
__global__ __launch_bounds__(256)
void router_kernel(const float* __restrict__ x, const float* __restrict__ gate_w,
                   float* __restrict__ probs, float* __restrict__ routef,
                   int* __restrict__ routei, int* __restrict__ list,
                   int* __restrict__ counts) {
  int warp = threadIdx.x >> 5, lane = threadIdx.x & 31;
  int t = blockIdx.x * 8 + warp;
  const float* xr = x + (size_t)t * DDIM;
  float p[NEXP];
#pragma unroll
  for (int e = 0; e < NEXP; ++e) p[e] = 0.0f;
  for (int d = lane; d < DDIM; d += 32) {
    float xv = xr[d];
#pragma unroll
    for (int e = 0; e < NEXP; ++e) p[e] += xv * gate_w[e * DDIM + d];
  }
#pragma unroll
  for (int e = 0; e < NEXP; ++e)
#pragma unroll
    for (int m = 16; m >= 1; m >>= 1) p[e] += __shfl_xor(p[e], m, 32);

  // softmax (deterministic, fixed order)
  float mx = p[0];
#pragma unroll
  for (int e = 1; e < NEXP; ++e) mx = fmaxf(mx, p[e]);
  float s = 0.0f;
#pragma unroll
  for (int e = 0; e < NEXP; ++e) { p[e] = expf(p[e] - mx); s += p[e]; }
  float inv = 1.0f / s;
#pragma unroll
  for (int e = 0; e < NEXP; ++e) p[e] *= inv;

  if (lane == 0) {
#pragma unroll
    for (int e = 0; e < NEXP; ++e) probs[t * NEXP + e] = p[e];
    // top-2, first-index tie-break like jax.lax.top_k
    int i0 = 0; float w0 = p[0];
#pragma unroll
    for (int e = 1; e < NEXP; ++e) if (p[e] > w0) { w0 = p[e]; i0 = e; }
    int i1 = -1; float w1 = -1.0f;
#pragma unroll
    for (int e = 0; e < NEXP; ++e) if (e != i0 && p[e] > w1) { w1 = p[e]; i1 = e; }
    float nrm = 1.0f / (w0 + w1 + 1e-9f);
    w0 *= nrm; w1 *= nrm;
    int s0 = atomicAdd(&counts[i0], 1); list[i0 * N_TOK + s0] = t;
    int s1 = atomicAdd(&counts[i1], 1); list[i1 * N_TOK + s1] = t;
    routef[t * 2 + 0] = w0; routef[t * 2 + 1] = w1;
    routei[t * 4 + 0] = i0; routei[t * 4 + 1] = i1;
    routei[t * 4 + 2] = s0; routei[t * 4 + 3] = s1;
  }
}

// ---------------- kernel 3: exclusive scan of expert counts ----------------
__global__ void offsets_kernel(const int* __restrict__ counts, int* __restrict__ offs) {
  if (threadIdx.x == 0) {
    int run = 0;
    for (int e = 0; e < NEXP; ++e) { offs[e] = run; run += counts[e]; }
  }
}

// ---------------- kernel 4: aux loss (deterministic fixed-tree reduction) ----
__global__ __launch_bounds__(256)
void aux_kernel(const float* __restrict__ probs, float* __restrict__ out_scalar) {
  __shared__ float sarr[NEXP];
  int warp = threadIdx.x >> 5, lane = threadIdx.x & 31;  // warp == expert
  float s = 0.0f;
  for (int t = lane; t < N_TOK; t += 32) s += probs[t * NEXP + warp];
#pragma unroll
  for (int m = 16; m >= 1; m >>= 1) s += __shfl_xor(s, m, 32);
  if (lane == 0) sarr[warp] = s;
  __syncthreads();
  if (threadIdx.x == 0) {
    float loss = 0.0f;
    for (int e = 0; e < NEXP; ++e) {
      float avg = sarr[e] * (1.0f / N_TOK);
      float d = avg - 1.0f / NEXP;
      loss += d * d;                       // mean over E of d^2, times E == sum d^2
    }
    out_scalar[0] = loss;
  }
}

// ---------------- kernel 5: fused expert FFN (bf16 WMMA) ----------------
// grid = (N/TM m-tiles, NEXP experts), 512 threads = 16 waves.
// Block computes TM=64 gathered tokens x full D output for one expert.
// Each wave: 4 M-subtiles x 4 N-tiles (16x16 each); B fragments reused 4x.
__global__ __launch_bounds__(512)
void ffn_kernel(const unsigned short* __restrict__ w1bf,   // (E,F,D) bf16
                const unsigned short* __restrict__ w2bf,   // (E,D,F) bf16
                const unsigned short* __restrict__ xbf,    // (N,D)   bf16
                const float* __restrict__ fc1_b,           // (E,F)
                const float* __restrict__ fc2_b,           // (E,D)
                const int* __restrict__ list,              // (E,N) token ids
                const int* __restrict__ counts,            // (E)
                const int* __restrict__ offs,              // (E) row offsets
                float* __restrict__ ypair)                 // (2N, D) f32
{
  int e = blockIdx.y;
  int cnt = counts[e];
  int m0 = blockIdx.x * TM;
  if (m0 >= cnt) return;
  int rem = cnt - m0; if (rem > TM) rem = TM;

  __shared__ unsigned short Xs[TM * XPITCH];  // 64 x 1024 bf16 token tile (132KB)
  __shared__ unsigned short Hs[TM * HPITCH];  // 64 x 256  bf16 hidden tile (34KB)

  int tid  = threadIdx.x;
  int wv   = tid >> 5;       // wave id 0..15
  int lane = tid & 31;
  int half = lane >> 4;      // 0 / 1
  int nn   = lane & 15;      // row (A) / column (B,C) index

  // ---- gather token tile into LDS ----
#pragma unroll
  for (int q = 0; q < 4; ++q) {
    int idx = tid + q * 512;          // 0..2047 (row, 32-chunk) pairs
    int r = idx >> 5;                 // row 0..63
    int c = (idx & 31) * 32;          // ushort column
    int tok = list[e * N_TOK + m0 + (r < rem ? r : 0)];
    const uint4* src = (const uint4*)(xbf + (size_t)tok * DDIM + c);
    uint4* dst = (uint4*)(&Xs[r * XPITCH + c]);
#pragma unroll
    for (int u = 0; u < 4; ++u) dst[u] = src[u];
  }
  __syncthreads();

  // per-wave output columns: d in [wv*64, wv*64+64), 4 n-tiles of 16
  const unsigned short* w2r[4];
#pragma unroll
  for (int t = 0; t < 4; ++t)
    w2r[t] = w2bf + ((size_t)e * DDIM + wv * 64 + t * 16 + nn) * (size_t)FDIM;

  v8f acc[MT][4];
#pragma unroll
  for (int s = 0; s < MT; ++s)
#pragma unroll
    for (int t = 0; t < 4; ++t) acc[s][t] = (v8f){0,0,0,0,0,0,0,0};

  for (int f0 = 0; f0 < FDIM; f0 += FB) {
    // ---- Phase A: H[0:64, wv*16 .. +16) = gelu(X @ W1^T + b1) ----
    int fcol = f0 + wv * 16 + nn;     // this lane's output column of H
    const unsigned short* w1row = w1bf + ((size_t)e * FDIM + fcol) * (size_t)DDIM;
    v8f hc[MT];
#pragma unroll
    for (int s = 0; s < MT; ++s) hc[s] = (v8f){0,0,0,0,0,0,0,0};
#pragma unroll 2
    for (int kc = 0; kc < DDIM; kc += 32) {
      const uint4* bp = (const uint4*)(w1row + kc + half * 16);
      v16bf bfrag = make_frag(bp[0], bp[1]);
      int ka = kc + half * 16;
#pragma unroll
      for (int s = 0; s < MT; ++s) {
        uint4 x0 = *(const uint4*)(&Xs[(s * 16 + nn) * XPITCH + ka]);
        uint4 x1 = *(const uint4*)(&Xs[(s * 16 + nn) * XPITCH + ka + 8]);
        hc[s] = __builtin_amdgcn_wmma_f32_16x16x32_bf16(
                   false, make_frag(x0, x1), false, bfrag, (short)0, hc[s],
                   false, false);
      }
    }
    float b1v = fc1_b[e * FDIM + fcol];
    __syncthreads();                  // previous Phase-B readers done
#pragma unroll
    for (int s = 0; s < MT; ++s)
#pragma unroll
      for (int v = 0; v < 8; ++v) {
        float xg = hc[s][v] + b1v;
        float g  = 0.5f * xg * (1.0f + erff(xg * 0.70710678118654752f));
        Hs[(s * 16 + v + 8 * half) * HPITCH + wv * 16 + nn] = f2bf(g);
      }
    __syncthreads();                  // H chunk visible to all waves

    // ---- Phase B: Y[0:64, wv*64 .. +64) += H_chunk @ W2^T ----
#pragma unroll
    for (int kc = 0; kc < FB; kc += 32) {
      int ka = kc + half * 16;
      v16bf af[MT];
#pragma unroll
      for (int s = 0; s < MT; ++s) {
        uint4 h0 = *(const uint4*)(&Hs[(s * 16 + nn) * HPITCH + ka]);
        uint4 h1 = *(const uint4*)(&Hs[(s * 16 + nn) * HPITCH + ka + 8]);
        af[s] = make_frag(h0, h1);
      }
      int koff = f0 + kc + half * 16;
#pragma unroll
      for (int t = 0; t < 4; ++t) {
        const uint4* bp = (const uint4*)(w2r[t] + koff);
        v16bf bfrag = make_frag(bp[0], bp[1]);
#pragma unroll
        for (int s = 0; s < MT; ++s)
          acc[s][t] = __builtin_amdgcn_wmma_f32_16x16x32_bf16(
                         false, af[s], false, bfrag, (short)0, acc[s][t],
                         false, false);
      }
    }
  }

  // ---- epilogue: add fc2 bias, store fp32 pair rows ----
  int prow0 = offs[e] + m0;
#pragma unroll
  for (int t = 0; t < 4; ++t) {
    int d = wv * 64 + t * 16 + nn;
    float b2v = fc2_b[e * DDIM + d];
#pragma unroll
    for (int s = 0; s < MT; ++s)
#pragma unroll
      for (int v = 0; v < 8; ++v) {
        int m = s * 16 + v + 8 * half;  // C/D layout: M = v + 8*(lane>>4), N = lane&15
        if (m < rem)
          ypair[(size_t)(prow0 + m) * DDIM + d] = acc[s][t][v] + b2v;
      }
  }
}

// ---------------- kernel 6: combine routed pair outputs ----------------
__global__ __launch_bounds__(256)
void combine_kernel(const float* __restrict__ ypair, const float* __restrict__ routef,
                    const int* __restrict__ routei, const int* __restrict__ offs,
                    float* __restrict__ out) {
  int gid = blockIdx.x * blockDim.x + threadIdx.x;       // one float4 each
  if (gid >= N_TOK * DDIM / 4) return;
  int t  = gid >> 8;                                     // 256 float4 per token
  int d  = (gid & 255) * 4;
  int i0 = routei[t * 4 + 0], i1 = routei[t * 4 + 1];
  int s0 = routei[t * 4 + 2], s1 = routei[t * 4 + 3];
  float w0 = routef[t * 2 + 0], w1 = routef[t * 2 + 1];
  const float4 y0 = *(const float4*)(ypair + (size_t)(offs[i0] + s0) * DDIM + d);
  const float4 y1 = *(const float4*)(ypair + (size_t)(offs[i1] + s1) * DDIM + d);
  float4 o;
  o.x = w0 * y0.x + w1 * y1.x;
  o.y = w0 * y0.y + w1 * y1.y;
  o.z = w0 * y0.z + w1 * y1.z;
  o.w = w0 * y0.w + w1 * y1.w;
  *(float4*)(out + (size_t)gid * 4) = o;
}

// ---------------- host side ----------------
extern "C" void kernel_launch(void* const* d_in, const int* in_sizes, int n_in,
                              void* d_out, int out_size, void* d_ws, size_t ws_size,
                              hipStream_t stream) {
  const float* x      = (const float*)d_in[0];
  const float* gate_w = (const float*)d_in[1];
  const float* fc1_w  = (const float*)d_in[2];
  const float* fc1_b  = (const float*)d_in[3];
  const float* fc2_w  = (const float*)d_in[4];
  const float* fc2_b  = (const float*)d_in[5];
  float* out = (float*)d_out;

  // workspace layout (all offsets 256B aligned)
  char* ws = (char*)d_ws;
  constexpr size_t SZ_W1 = (size_t)NEXP * FDIM * DDIM * 2;  // 64 MB bf16 fc1
  constexpr size_t SZ_W2 = (size_t)NEXP * DDIM * FDIM * 2;  // 64 MB bf16 fc2
  constexpr size_t SZ_X  = (size_t)N_TOK * DDIM * 2;        // 16 MB bf16 x
  constexpr size_t SZ_Y  = (size_t)2 * N_TOK * DDIM * 4;    // 64 MB pair outputs
  constexpr size_t SZ_P  = (size_t)N_TOK * NEXP * 4;        // probs
  constexpr size_t SZ_RF = (size_t)N_TOK * 2 * 4;           // routing weights
  constexpr size_t SZ_RI = (size_t)N_TOK * 4 * 4;           // routing ints
  constexpr size_t SZ_L  = (size_t)NEXP * N_TOK * 4;        // expert token lists
  unsigned short* w1bf  = (unsigned short*)(ws);
  unsigned short* w2bf  = (unsigned short*)(ws + SZ_W1);
  unsigned short* xbf   = (unsigned short*)(ws + SZ_W1 + SZ_W2);
  float* ypair          = (float*)(ws + SZ_W1 + SZ_W2 + SZ_X);
  float* probs          = (float*)(ws + SZ_W1 + SZ_W2 + SZ_X + SZ_Y);
  float* routef         = (float*)(ws + SZ_W1 + SZ_W2 + SZ_X + SZ_Y + SZ_P);
  int*   routei         = (int*)  (ws + SZ_W1 + SZ_W2 + SZ_X + SZ_Y + SZ_P + SZ_RF);
  int*   list           = (int*)  (ws + SZ_W1 + SZ_W2 + SZ_X + SZ_Y + SZ_P + SZ_RF + SZ_RI);
  int*   counts         = (int*)  (ws + SZ_W1 + SZ_W2 + SZ_X + SZ_Y + SZ_P + SZ_RF + SZ_RI + SZ_L);
  int*   offs           = counts + NEXP;
  (void)in_sizes; (void)n_in; (void)out_size; (void)ws_size;

  init_kernel<<<1, 32, 0, stream>>>(counts);

  // f32 -> bf16 (weights become L2-resident working set: 128 MB < 192 MB L2)
  cvt_bf16_kernel<<<(NEXP * FDIM * DDIM / 4 + 255) / 256, 256, 0, stream>>>(fc1_w, w1bf, NEXP * FDIM * DDIM / 4);
  cvt_bf16_kernel<<<(NEXP * DDIM * FDIM / 4 + 255) / 256, 256, 0, stream>>>(fc2_w, w2bf, NEXP * DDIM * FDIM / 4);
  cvt_bf16_kernel<<<(N_TOK * DDIM / 4 + 255) / 256, 256, 0, stream>>>(x, xbf, N_TOK * DDIM / 4);

  router_kernel<<<N_TOK / 8, 256, 0, stream>>>(x, gate_w, probs, routef, routei, list, counts);
  offsets_kernel<<<1, 1, 0, stream>>>(counts, offs);
  aux_kernel<<<1, 256, 0, stream>>>(probs, out + (size_t)N_TOK * DDIM);

  dim3 ffn_grid(N_TOK / TM, NEXP);   // worst case: every token routed to one expert
  ffn_kernel<<<ffn_grid, 512, 0, stream>>>(w1bf, w2bf, xbf, fc1_b, fc2_b,
                                           list, counts, offs, ypair);

  combine_kernel<<<(N_TOK * DDIM / 4) / 256, 256, 0, stream>>>(ypair, routef, routei, offs, out);
}
